// TRGA_Block_5420248728085
// MI455X (gfx1250) — matrix operand order repaired and checked by reading, not compiled, not run on hardware
//
#include <hip/hip_runtime.h>
#include <hip/hip_bf16.h>

typedef __attribute__((ext_vector_type(16))) _Float16 v16h;
typedef __attribute__((ext_vector_type(8)))  float    v8f;

union AF { v16h h; unsigned u[8]; };

#define BATCH  8
#define CH     128
#define NPT    1000
#define NTOK   1001
#define NHEADS 4
#define DHEAD  32
#define EPS_   1e-5f

__device__ __forceinline__ unsigned pkh(float a, float b) {
    union { _Float16 h[2]; unsigned u; } x;
    x.h[0] = (_Float16)a; x.h[1] = (_Float16)b;
    return x.u;
}

// ---------------------------------------------------------------------------
// WMMA GEMM:  C[b] = act( A[b] (MxK) * B[b] (KxN) + bias + res )
// 256 threads = 8 waves, block tile 32(M) x 64(N), K step 32.
// Software-pipelined: tile k+1 is register-prefetched while tile k computes;
// double-buffered LDS -> single barrier per K step.
// biasMode: 0 none, 1 per-col, 2 per-row.  act: 0 none, 1 relu, 2 gelu(exact)
// ---------------------------------------------------------------------------
__global__ __launch_bounds__(256)
void k_gemm(const float* __restrict__ A, const float* __restrict__ Bm,
            const float* __restrict__ bias, int biasMode,
            const float* __restrict__ res, float* __restrict__ C,
            int M, int N, int K,
            long sAb, long sBb, long sCb,
            int lda, int ldb, int ldc, int act)
{
    __shared__ unsigned sAp[2][32][17];   // [buf][row][K-pair] (pair = k,k+1 lo/hi)
    __shared__ unsigned sBp[2][16][65];   // [buf][K-pair][col]
    const int bz = blockIdx.z;
    const float* Ab = A  + (long)bz * sAb;
    const float* Bb = Bm + (long)bz * sBb;
    const int mBase = blockIdx.y * 32;
    const int nBase = blockIdx.x * 64;
    const int tid  = threadIdx.x;
    const int lane = tid & 31;
    const int wave = tid >> 5;
    const int hf   = lane >> 4;
    const int l15  = lane & 15;
    const int mSub = (wave >> 2) * 16;
    const int nSub = (wave & 3) * 16;

    // fixed per-thread staging positions (cover each tile exactly once)
    const int rA = tid >> 3, cA = (tid & 7) * 4;     // A: 32 rows x 32 k
    const int pjB = tid >> 4, cB = (tid & 15) * 4;   // B: 16 k-pairs x 64 cols
    const bool edgeM = (mBase + 32 > M);
    const bool edgeN = (nBase + 64 > N);

    float4 a, b0, b1;
    auto pref = [&](int k0) {
        const bool edgeK = (k0 + 32 > K);
        if (!edgeM && !edgeK) {
            a = *(const float4*)(Ab + (long)(mBase + rA) * lda + k0 + cA);
        } else {
            int gr = mBase + rA;
            int cr = gr < M ? gr : M - 1;
            const float* rp = Ab + (long)cr * lda;
            auto g = [&](int k) {
                int ck = k < K ? k : K - 1;
                float t = rp[ck];
                return (gr < M && k < K) ? t : 0.f;
            };
            a.x = g(k0 + cA); a.y = g(k0 + cA + 1);
            a.z = g(k0 + cA + 2); a.w = g(k0 + cA + 3);
        }
        if (!edgeN && !edgeK) {
            b0 = *(const float4*)(Bb + (long)(k0 + 2 * pjB)     * ldb + nBase + cB);
            b1 = *(const float4*)(Bb + (long)(k0 + 2 * pjB + 1) * ldb + nBase + cB);
        } else {
            int kr0 = k0 + 2 * pjB, kr1 = kr0 + 1;
            const float* rp0 = Bb + (long)(kr0 < K ? kr0 : K - 1) * ldb;
            const float* rp1 = Bb + (long)(kr1 < K ? kr1 : K - 1) * ldb;
            auto g = [&](const float* rp, int kr, int c) {
                int cc = c < N ? c : N - 1;
                float t = rp[cc];
                return (kr < K && c < N) ? t : 0.f;
            };
            b0.x = g(rp0, kr0, nBase + cB);     b0.y = g(rp0, kr0, nBase + cB + 1);
            b0.z = g(rp0, kr0, nBase + cB + 2); b0.w = g(rp0, kr0, nBase + cB + 3);
            b1.x = g(rp1, kr1, nBase + cB);     b1.y = g(rp1, kr1, nBase + cB + 1);
            b1.z = g(rp1, kr1, nBase + cB + 2); b1.w = g(rp1, kr1, nBase + cB + 3);
        }
    };

    pref(0);
    int p = 0;
    v8f acc = {};
    for (int k0 = 0; k0 < K; k0 += 32) {
        sAp[p][rA][(cA >> 1)]     = pkh(a.x, a.y);
        sAp[p][rA][(cA >> 1) + 1] = pkh(a.z, a.w);
        sBp[p][pjB][cB]     = pkh(b0.x, b1.x);
        sBp[p][pjB][cB + 1] = pkh(b0.y, b1.y);
        sBp[p][pjB][cB + 2] = pkh(b0.z, b1.z);
        sBp[p][pjB][cB + 3] = pkh(b0.w, b1.w);
        if (k0 + 32 < K) pref(k0 + 32);        // overlap with barrier + WMMA
        __syncthreads();
        AF af, bf;
        #pragma unroll
        for (int vv = 0; vv < 8; ++vv) {
            int pi = (vv < 4) ? (hf * 4 + vv) : (8 + hf * 4 + (vv - 4));
            af.u[vv] = sAp[p][mSub + l15][pi];
            bf.u[vv] = sBp[p][pi][nSub + l15];
        }
        acc = __builtin_amdgcn_wmma_f32_16x16x32_f16(false, af.h, false, bf.h,
                                                     (short)0, acc, false, false);
        p ^= 1;
    }
    const float* Rb = res ? res + (long)bz * sCb : (const float*)0;
    float* Cb = C + (long)bz * sCb;
    const int col = nBase + nSub + l15;
    float bcol = (biasMode == 1 && bias && col < N) ? bias[col] : 0.f;
    #pragma unroll
    for (int r = 0; r < 8; ++r) {
        int row = mBase + mSub + r + 8 * hf;
        if (row < M && col < N) {
            float x = acc[r] + bcol;
            if (biasMode == 2 && bias) x += bias[row];
            if (Rb) x += Rb[(long)row * ldc + col];
            if (act == 1)      x = fmaxf(x, 0.f);
            else if (act == 2) x = 0.5f * x * (1.f + erff(x * 0.70710678118f));
            Cb[(long)row * ldc + col] = x;
        }
    }
}

// ---------------------------------------------------------------------------
// Flash attention: 128 threads = 4 waves, 64 query rows per block (16/wave).
// K/V tiles staged in double-buffered packed-pair LDS shared by all 4 waves;
// next tile register-prefetched during online softmax; ONE barrier per tile.
// P's C-frag -> A-frag re-layout uses per-wave LDS (same-wave DS ordering).
// ---------------------------------------------------------------------------
__global__ __launch_bounds__(128)
void k_attn(const float* __restrict__ qkv, float* __restrict__ O)
{
    __shared__ unsigned sKp[2][16][33];     // [buf][d-pair][j]
    __shared__ unsigned sVp[2][16][33];     // [buf][j-pair][d]
    __shared__ _Float16 sP[4][16][34];      // per-wave P re-layout buffer
    const int bh = blockIdx.y;
    const int b = bh / NHEADS, h = bh % NHEADS;
    const int tid = threadIdx.x;
    const int lane = tid & 31, wave = tid >> 5;
    const int hf = lane >> 4, l15 = lane & 15;
    const int rBase = blockIdx.x * 64 + wave * 16;
    const int ldq = 3 * CH;
    const float* Q  = qkv + (long)b * NTOK * ldq + h * DHEAD;
    const float* Kp = Q + CH;
    const float* Vp = Q + 2 * CH;
    const float scale = rsqrtf((float)DHEAD);

    AF qf;
    {
        int row = rBase + l15; if (row >= NTOK) row = NTOK - 1;
        const float* qr = Q + (long)row * ldq;
        const float4 a0 = *(const float4*)(qr + hf * 8);
        const float4 a1 = *(const float4*)(qr + hf * 8 + 4);
        const float4 a2 = *(const float4*)(qr + 16 + hf * 8);
        const float4 a3 = *(const float4*)(qr + 16 + hf * 8 + 4);
        qf.u[0] = pkh(a0.x * scale, a0.y * scale);
        qf.u[1] = pkh(a0.z * scale, a0.w * scale);
        qf.u[2] = pkh(a1.x * scale, a1.y * scale);
        qf.u[3] = pkh(a1.z * scale, a1.w * scale);
        qf.u[4] = pkh(a2.x * scale, a2.y * scale);
        qf.u[5] = pkh(a2.z * scale, a2.w * scale);
        qf.u[6] = pkh(a3.x * scale, a3.y * scale);
        qf.u[7] = pkh(a3.z * scale, a3.w * scale);
    }

    // fixed per-thread staging positions
    const int jK0 = tid >> 3, jK1 = 16 + (tid >> 3), dK = (tid & 7) * 4;
    const int pjV = tid >> 3, dV = (tid & 7) * 4;

    float4 rK0, rK1, rV0, rV1;
    auto prefKV = [&](int j0) {
        int a0 = j0 + jK0; if (a0 >= NTOK) a0 = NTOK - 1;
        int a1 = j0 + jK1; if (a1 >= NTOK) a1 = NTOK - 1;
        rK0 = *(const float4*)(Kp + (long)a0 * ldq + dK);
        rK1 = *(const float4*)(Kp + (long)a1 * ldq + dK);
        int v0 = j0 + 2 * pjV;     if (v0 >= NTOK) v0 = NTOK - 1;
        int v1 = j0 + 2 * pjV + 1; if (v1 >= NTOK) v1 = NTOK - 1;
        rV0 = *(const float4*)(Vp + (long)v0 * ldq + dV);
        rV1 = *(const float4*)(Vp + (long)v1 * ldq + dV);
    };

    float m[8], l[8];
    #pragma unroll
    for (int r = 0; r < 8; ++r) { m[r] = -3.0e30f; l[r] = 0.f; }
    v8f Oa = {}, Ob = {};

    prefKV(0);
    int p = 0;
    for (int j0 = 0; j0 < NTOK; j0 += 32) {
        sKp[p][(dK >> 1)][jK0]     = pkh(rK0.x, rK0.y);
        sKp[p][(dK >> 1) + 1][jK0] = pkh(rK0.z, rK0.w);
        sKp[p][(dK >> 1)][jK1]     = pkh(rK1.x, rK1.y);
        sKp[p][(dK >> 1) + 1][jK1] = pkh(rK1.z, rK1.w);
        sVp[p][pjV][dV]     = pkh(rV0.x, rV1.x);
        sVp[p][pjV][dV + 1] = pkh(rV0.y, rV1.y);
        sVp[p][pjV][dV + 2] = pkh(rV0.z, rV1.z);
        sVp[p][pjV][dV + 3] = pkh(rV0.w, rV1.w);
        if (j0 + 32 < NTOK) prefKV(j0 + 32);   // overlap with softmax below
        __syncthreads();

        AF kf0, kf1;
        #pragma unroll
        for (int vv = 0; vv < 8; ++vv) {
            int pi = (vv < 4) ? (hf * 4 + vv) : (8 + hf * 4 + (vv - 4));
            kf0.u[vv] = sKp[p][pi][l15];
            kf1.u[vv] = sKp[p][pi][16 + l15];
        }
        v8f S0 = {}, S1 = {};
        S0 = __builtin_amdgcn_wmma_f32_16x16x32_f16(false, qf.h, false, kf0.h, (short)0, S0, false, false);
        S1 = __builtin_amdgcn_wmma_f32_16x16x32_f16(false, qf.h, false, kf1.h, (short)0, S1, false, false);

        const int c0 = j0 + l15, c1 = j0 + 16 + l15;
        float pa[8], pb[8];
        #pragma unroll
        for (int r = 0; r < 8; ++r) {
            float s0 = (c0 < NTOK) ? S0[r] : -3.0e30f;
            float s1 = (c1 < NTOK) ? S1[r] : -3.0e30f;
            float mt = fmaxf(s0, s1);
            #pragma unroll
            for (int off = 1; off < 16; off <<= 1) mt = fmaxf(mt, __shfl_xor(mt, off, 32));
            float mn = fmaxf(m[r], mt);
            float alpha = __expf(m[r] - mn);
            float e0 = (c0 < NTOK) ? __expf(s0 - mn) : 0.f;
            float e1 = (c1 < NTOK) ? __expf(s1 - mn) : 0.f;
            float ls = e0 + e1;
            #pragma unroll
            for (int off = 1; off < 16; off <<= 1) ls += __shfl_xor(ls, off, 32);
            l[r] = l[r] * alpha + ls;
            m[r] = mn;
            Oa[r] *= alpha; Ob[r] *= alpha;
            pa[r] = e0; pb[r] = e1;
        }
        #pragma unroll
        for (int r = 0; r < 8; ++r) {           // C-layout -> per-wave LDS
            sP[wave][r + 8 * hf][l15]      = (_Float16)pa[r];
            sP[wave][r + 8 * hf][16 + l15] = (_Float16)pb[r];
        }
        // same-wave DS ordering guarantees the stores above are visible to the
        // loads below; compiler inserts the dscnt wait.
        AF pf, vf0, vf1;
        #pragma unroll
        for (int vv = 0; vv < 8; ++vv) {        // LDS -> A-layout (pairs)
            int kk = (vv < 4) ? (hf * 8 + 2 * vv) : (16 + hf * 8 + 2 * (vv - 4));
            pf.u[vv]  = *(const unsigned*)&sP[wave][l15][kk];
            vf0.u[vv] = sVp[p][kk >> 1][l15];
            vf1.u[vv] = sVp[p][kk >> 1][16 + l15];
        }
        Oa = __builtin_amdgcn_wmma_f32_16x16x32_f16(false, pf.h, false, vf0.h, (short)0, Oa, false, false);
        Ob = __builtin_amdgcn_wmma_f32_16x16x32_f16(false, pf.h, false, vf1.h, (short)0, Ob, false, false);
        p ^= 1;
    }
    float* Op = O + (long)b * NTOK * CH + h * DHEAD;
    #pragma unroll
    for (int r = 0; r < 8; ++r) {
        int row = rBase + r + 8 * hf;
        if (row < NTOK) {
            float inv = 1.f / l[r];
            Op[(long)row * CH + l15]      = Oa[r] * inv;
            Op[(long)row * CH + 16 + l15] = Ob[r] * inv;
        }
    }
}

// ---------------------------------------------------------------------------
// small utility kernels
// ---------------------------------------------------------------------------
__device__ __forceinline__ float blockReduceSum(float v, float* sh) {
    int tid = threadIdx.x;
    sh[tid] = v; __syncthreads();
    for (int s = blockDim.x >> 1; s > 0; s >>= 1) {
        if (tid < s) sh[tid] += sh[tid + s];
        __syncthreads();
    }
    float r = sh[0]; __syncthreads();
    return r;
}

__global__ void k_transpose(const float* __restrict__ in, float* __restrict__ out,
                            int R, int Cd, long sIb, long sOb)
{   // out[b][c][r] = in[b][r][c]
    __shared__ float tile[32][33];
    const float* ib = in + (long)blockIdx.z * sIb;
    float* ob = out + (long)blockIdx.z * sOb;
    int r0 = blockIdx.y * 32, c0 = blockIdx.x * 32;
    int tx = threadIdx.x, ty = threadIdx.y;
    for (int i = ty; i < 32; i += 8) {
        int r = r0 + i, c = c0 + tx;
        tile[i][tx] = (r < R && c < Cd) ? ib[(long)r * Cd + c] : 0.f;
    }
    __syncthreads();
    for (int i = ty; i < 32; i += 8) {
        int c = c0 + i, r = r0 + tx;
        if (c < Cd && r < R) ob[(long)c * R + r] = tile[tx][i];
    }
}

__global__ void k_set_cls(float* t, const float* cls) {
    t[(long)blockIdx.x * NTOK * CH + threadIdx.x] = cls[threadIdx.x];
}

__global__ __launch_bounds__(256)
void k_layernorm(const float* __restrict__ x, const float* __restrict__ g,
                 const float* __restrict__ bt, float* __restrict__ y, long rows)
{   // one wave per row, C = 128, float4 + shuffle reduction
    long row = (long)blockIdx.x * 8 + (threadIdx.x >> 5);
    if (row >= rows) return;
    int lane = threadIdx.x & 31;
    const float4 v = *(const float4*)(x + row * CH + lane * 4);
    float s = v.x + v.y + v.z + v.w;
    float q = v.x * v.x + v.y * v.y + v.z * v.z + v.w * v.w;
    #pragma unroll
    for (int off = 1; off < 32; off <<= 1) {
        s += __shfl_xor(s, off, 32);
        q += __shfl_xor(q, off, 32);
    }
    float mean = s / CH;
    float r = rsqrtf(q / CH - mean * mean + EPS_);
    const float4 gg = *(const float4*)(g + lane * 4);
    const float4 bb = *(const float4*)(bt + lane * 4);
    float4 o;
    o.x = (v.x - mean) * r * gg.x + bb.x;
    o.y = (v.y - mean) * r * gg.y + bb.y;
    o.z = (v.z - mean) * r * gg.z + bb.z;
    o.w = (v.w - mean) * r * gg.w + bb.w;
    *(float4*)(y + row * CH + lane * 4) = o;
}

__global__ void k_instnorm(const float* __restrict__ x, float* __restrict__ y, int N)
{   // one block per (b,c), normalize over N
    long bc = blockIdx.x;
    const float* xr = x + bc * (long)N;
    float* yr = y + bc * (long)N;
    __shared__ float sh[256];
    float s = 0.f, q = 0.f;
    for (int i = threadIdx.x; i < N; i += 256) { float v = xr[i]; s += v; q += v * v; }
    s = blockReduceSum(s, sh);
    q = blockReduceSum(q, sh);
    float mean = s / N, var = q / N - mean * mean;
    float r = rsqrtf(var + EPS_);
    for (int i = threadIdx.x; i < N; i += 256) yr[i] = (xr[i] - mean) * r;
}

__global__ void k_bn_stats(const float* __restrict__ x, float* __restrict__ mv,
                           int Bn, int Cb, int N)
{   // one block per channel; stats over (B, N)
    int c = blockIdx.x;
    __shared__ float sh[256];
    float s = 0.f, q = 0.f;
    for (int b = 0; b < Bn; ++b) {
        const float* xr = x + ((long)b * Cb + c) * N;
        for (int n = threadIdx.x; n < N; n += 256) { float v = xr[n]; s += v; q += v * v; }
    }
    s = blockReduceSum(s, sh);
    q = blockReduceSum(q, sh);
    if (threadIdx.x == 0) {
        float total = (float)Bn * N;
        float mean = s / total;
        mv[c] = mean;
        mv[Cb + c] = q / total - mean * mean;
    }
}

__global__ void k_bn_apply(const float* __restrict__ x, const float* mv,
                           const float* g, const float* bt, float* __restrict__ y,
                           int Cb, int N, int relu, long total)
{
    long i = (long)blockIdx.x * blockDim.x + threadIdx.x;
    if (i >= total) return;
    int c = (int)((i / N) % Cb);
    float r = rsqrtf(mv[Cb + c] + EPS_);
    float o = (x[i] - mv[c]) * r * g[c] + bt[c];
    if (relu) o = fmaxf(o, 0.f);
    y[i] = o;
}

__global__ void k_add(const float* a, const float* b, float* y, int relu, long total) {
    long i = (long)blockIdx.x * blockDim.x + threadIdx.x;
    if (i >= total) return;
    float v = a[i] + b[i];
    y[i] = relu ? fmaxf(v, 0.f) : v;
}

__global__ void k_rowdot(const float* __restrict__ x, const float* __restrict__ w,
                         const float* bias, float* __restrict__ out, int Cb, int N, long total)
{   // out[b,n] = sum_c w[c] * x[b,c,n] + bias[0]
    long i = (long)blockIdx.x * blockDim.x + threadIdx.x;
    if (i >= total) return;
    long b = i / N, n = i % N;
    float s = bias[0];
    for (int c = 0; c < Cb; ++c) s += w[c] * x[((long)b * Cb + c) * N + n];
    out[i] = s;
}

// ----- cgns (rank-1 + identity structure) -----------------------------------
__global__ void k_cgns_w(const float* __restrict__ w0, float* wr, float* d, float* dwr, int N)
{
    int b = blockIdx.x;
    __shared__ float sh[256];
    float s = 0.f;
    for (int i = threadIdx.x; i < N; i += 256) {
        float t = fmaxf(tanhf(w0[(long)b * N + i]), 0.f);
        wr[(long)b * N + i] = t;
        s += t;
    }
    float S = blockReduceSum(s, sh);
    for (int i = threadIdx.x; i < N; i += 256) {
        float t = wr[(long)b * N + i];
        float dd = rsqrtf(t * S + 1.f);     // A.sum(-1)[n] = wr[n]*S + 1
        d[(long)b * N + i] = dd;
        dwr[(long)b * N + i] = dd * t;
    }
}

__global__ void k_cgns_dots(const float* __restrict__ x, const float* wr, const float* dwr,
                            float* dot0, float* dot1, int Cb, int N)
{
    long bc = blockIdx.x;
    long b = bc / Cb;
    const float* xr = x + bc * (long)N;
    __shared__ float sh[256];
    float s0 = 0.f, s1 = 0.f;
    for (int i = threadIdx.x; i < N; i += 256) {
        float v = xr[i];
        s0 += v * wr[b * N + i];
        s1 += v * dwr[b * N + i];
    }
    s0 = blockReduceSum(s0, sh);
    s1 = blockReduceSum(s1, sh);
    if (threadIdx.x == 0) { dot0[bc] = s0; dot1[bc] = s1; }
}

__global__ void k_cgns_combine(const float* __restrict__ x, const float* wr, const float* d,
                               const float* dwr, const float* dot0, const float* dot1,
                               float* __restrict__ out, int Cb, int N, long total)
{
    long i = (long)blockIdx.x * blockDim.x + threadIdx.x;
    if (i >= total) return;
    long n = i % N;
    long c = (i / N) % Cb;
    long b = i / ((long)Cb * N);
    float xv = x[i];
    out[((long)b * 2 * Cb + c) * N + n] = dot0[b * Cb + c] * wr[b * N + n] + xv;
    float dd = d[b * N + n];
    out[((long)b * 2 * Cb + Cb + c) * N + n] = dwr[b * N + n] * dot1[b * Cb + c] + dd * dd * xv;
}

// ----- weighted 8-points ----------------------------------------------------
__global__ void k_xwx(const float* __restrict__ xs, const float* __restrict__ logits,
                      float* __restrict__ XwX, int N)
{
    int b = blockIdx.x;
    __shared__ float sh[256];
    float acc[45];
    #pragma unroll
    for (int s = 0; s < 45; ++s) acc[s] = 0.f;
    for (int n = threadIdx.x; n < N; n += 256) {
        const float4 p = *(const float4*)(xs + ((long)b * N + n) * 4);
        float u = p.x, vv = p.y, up = p.z, vp = p.w;
        float w = fmaxf(tanhf(logits[(long)b * N + n]), 0.f);
        float X[9] = { up * u, up * vv, up, vp * u, vp * vv, vp, u, vv, 1.f };
        int idx = 0;
        #pragma unroll
        for (int i = 0; i < 9; ++i)
            #pragma unroll
            for (int j = i; j < 9; ++j)
                acc[idx++] += w * X[i] * X[j];
    }
    for (int s = 0; s < 45; ++s) {
        float r = blockReduceSum(acc[s], sh);
        if (threadIdx.x == 0) XwX[b * 45 + s] = r;
    }
}

__global__ void k_eigh9(const float* __restrict__ XwX45, float* __restrict__ ehat, int Bn)
{
    int b = threadIdx.x;
    if (b >= Bn) return;
    float Am[9][9], Vm[9][9];
    int idx = 0;
    for (int i = 0; i < 9; ++i)
        for (int j = i; j < 9; ++j) {
            float v = XwX45[b * 45 + idx++];
            Am[i][j] = v; Am[j][i] = v;
        }
    for (int i = 0; i < 9; ++i)
        for (int j = 0; j < 9; ++j) Vm[i][j] = (i == j) ? 1.f : 0.f;
    for (int sweep = 0; sweep < 30; ++sweep) {
        for (int p = 0; p < 8; ++p)
            for (int q = p + 1; q < 9; ++q) {
                float apq = Am[p][q];
                if (fabsf(apq) < 1e-22f) continue;
                float theta = 0.5f * (Am[q][q] - Am[p][p]) / apq;
                float t = copysignf(1.f, theta) / (fabsf(theta) + sqrtf(theta * theta + 1.f));
                float c = rsqrtf(t * t + 1.f);
                float s = t * c;
                for (int k = 0; k < 9; ++k) {
                    float akp = Am[k][p], akq = Am[k][q];
                    Am[k][p] = c * akp - s * akq;
                    Am[k][q] = s * akp + c * akq;
                }
                for (int k = 0; k < 9; ++k) {
                    float apk = Am[p][k], aqk = Am[q][k];
                    Am[p][k] = c * apk - s * aqk;
                    Am[q][k] = s * apk + c * aqk;
                }
                for (int k = 0; k < 9; ++k) {
                    float vkp = Vm[k][p], vkq = Vm[k][q];
                    Vm[k][p] = c * vkp - s * vkq;
                    Vm[k][q] = s * vkp + c * vkq;
                }
            }
    }
    int mi = 0; float mv = Am[0][0];
    for (int i = 1; i < 9; ++i) if (Am[i][i] < mv) { mv = Am[i][i]; mi = i; }
    float nrm = 0.f;
    for (int i = 0; i < 9; ++i) nrm += Vm[i][mi] * Vm[i][mi];
    nrm = rsqrtf(nrm + 1e-30f);
    for (int i = 0; i < 9; ++i) ehat[b * 9 + i] = Vm[i][mi] * nrm;
}

__global__ void k_episym(const float* __restrict__ xs, const float* __restrict__ e,
                         float* __restrict__ out, int N, long total)
{
    long i = (long)blockIdx.x * blockDim.x + threadIdx.x;
    if (i >= total) return;
    long b = i / N;
    const float4 p = *(const float4*)(xs + i * 4);
    float x1x = p.x, x1y = p.y, x2x = p.z, x2y = p.w;
    const float* F = e + b * 9;
    float Fx0 = F[0] * x1x + F[1] * x1y + F[2];
    float Fx1 = F[3] * x1x + F[4] * x1y + F[5];
    float Fx2 = F[6] * x1x + F[7] * x1y + F[8];
    float Ft0 = F[0] * x2x + F[3] * x2y + F[6];
    float Ft1 = F[1] * x2x + F[4] * x2y + F[7];
    float d = x2x * Fx0 + x2y * Fx1 + Fx2;
    out[i] = d * d * (1.f / (Fx0 * Fx0 + Fx1 * Fx1 + 1e-15f)
                    + 1.f / (Ft0 * Ft0 + Ft1 * Ft1 + 1e-15f));
}

// ---------------------------------------------------------------------------
// host-side orchestration
// ---------------------------------------------------------------------------
struct LayerP { const float *b1,*b2,*bo,*ln1_b,*ln1_g,*ln2_b,*ln2_g,*w1,*w2,*wo,*wqkv; };
struct TransP { const float *be,*cls; LayerP L[2]; const float *we; };
struct ResP   { const float *b1,*b2,*be1,*be2,*g1,*g2,*rb,*rw,*w1,*w2; };

struct Ws {
    float *bufA, *bufB, *bufC, *bufD;
    float *t, *tn, *qkv, *attnO, *h1;
    float *mv, *wr, *dv, *dwr, *dot0, *dot1, *w0l, *xwx;
};

static void gemm(hipStream_t s, const float* A, const float* Bm,
                 const float* bias, int biasMode, const float* res, float* C,
                 int M, int N, int K, long sAb, long sBb, long sCb,
                 int lda, int ldb, int ldc, int act, int batch)
{
    dim3 g((N + 63) / 64, (M + 31) / 32, batch);
    k_gemm<<<g, 256, 0, s>>>(A, Bm, bias, biasMode, res, C,
                             M, N, K, sAb, sBb, sCb, lda, ldb, ldc, act);
}

static void transpose(hipStream_t s, const float* in, float* out,
                      int R, int Cd, long sIb, long sOb, int batch)
{
    dim3 g((Cd + 31) / 32, (R + 31) / 32, batch);
    k_transpose<<<g, dim3(32, 8), 0, s>>>(in, out, R, Cd, sIb, sOb);
}

static void layernorm(hipStream_t s, const float* x, const float* g,
                      const float* bt, float* y, long rows)
{
    k_layernorm<<<(int)((rows + 7) / 8), 256, 0, s>>>(x, g, bt, y, rows);
}

// transm: in (B,CH,NPT) channel-major -> out (B,CH,NPT)
static void run_transm(hipStream_t s, const Ws& w, const float* in, float* out, const TransP& tp)
{
    transpose(s, in, w.bufC, CH, NPT, (long)CH * NPT, (long)NPT * CH, BATCH);
    gemm(s, w.bufC, tp.we, tp.be, 1, 0, w.t + CH,
         NPT, CH, CH, (long)NPT * CH, 0, (long)NTOK * CH, CH, CH, CH, 0, BATCH);
    k_set_cls<<<BATCH, CH, 0, s>>>(w.t, tp.cls);

    for (int li = 0; li < 2; ++li) {
        const LayerP& L = tp.L[li];
        layernorm(s, w.t, L.ln1_g, L.ln1_b, w.tn, (long)BATCH * NTOK);
        gemm(s, w.tn, L.wqkv, 0, 0, 0, w.qkv,
             NTOK, 3 * CH, CH, (long)NTOK * CH, 0, (long)NTOK * 3 * CH, CH, 3 * CH, 3 * CH, 0, BATCH);
        {
            dim3 g((NTOK + 63) / 64, BATCH * NHEADS);
            k_attn<<<g, 128, 0, s>>>(w.qkv, w.attnO);
        }
        gemm(s, w.attnO, L.wo, L.bo, 1, w.t, w.t,
             NTOK, CH, CH, (long)NTOK * CH, 0, (long)NTOK * CH, CH, CH, CH, 0, BATCH);
        layernorm(s, w.t, L.ln2_g, L.ln2_b, w.tn, (long)BATCH * NTOK);
        gemm(s, w.tn, L.w1, L.b1, 1, 0, w.h1,
             NTOK, CH, CH, (long)NTOK * CH, 0, (long)NTOK * CH, CH, CH, CH, 2 /*gelu*/, BATCH);
        gemm(s, w.h1, L.w2, L.b2, 1, w.t, w.t,
             NTOK, CH, CH, (long)NTOK * CH, 0, (long)NTOK * CH, CH, CH, CH, 0, BATCH);
    }
    transpose(s, w.t + CH, out, NPT, CH, (long)NTOK * CH, (long)CH * NPT, BATCH);
}

// resnet block on channel-major tensors: x (B,ci,NPT) -> out (B,co,NPT)
static void run_resnet(hipStream_t s, const Ws& w, const float* x, float* out,
                       float* tmp1, float* tmp2, const ResP& rp, int ci, int co)
{
    long tot = (long)BATCH * co * NPT;
    int eb = (int)((tot + 255) / 256);
    gemm(s, rp.rw, x, rp.rb, 2, 0, out,
         co, NPT, ci, 0, (long)ci * NPT, (long)co * NPT, ci, NPT, NPT, 0, BATCH);
    gemm(s, rp.w1, x, rp.b1, 2, 0, tmp1,
         co, NPT, ci, 0, (long)ci * NPT, (long)co * NPT, ci, NPT, NPT, 0, BATCH);
    k_instnorm<<<BATCH * co, 256, 0, s>>>(tmp1, tmp1, NPT);
    k_bn_stats<<<co, 256, 0, s>>>(tmp1, w.mv, BATCH, co, NPT);
    k_bn_apply<<<eb, 256, 0, s>>>(tmp1, w.mv, rp.g1, rp.be1, tmp1, co, NPT, 1, tot);
    gemm(s, rp.w2, tmp1, rp.b2, 2, 0, tmp2,
         co, NPT, co, 0, (long)co * NPT, (long)co * NPT, co, NPT, NPT, 0, BATCH);
    k_instnorm<<<BATCH * co, 256, 0, s>>>(tmp2, tmp2, NPT);
    k_bn_stats<<<co, 256, 0, s>>>(tmp2, w.mv, BATCH, co, NPT);
    k_bn_apply<<<eb, 256, 0, s>>>(tmp2, w.mv, rp.g2, rp.be2, tmp2, co, NPT, 0, tot);
    k_add<<<eb, 256, 0, s>>>(tmp2, out, out, 1, tot);
}

extern "C" void kernel_launch(void* const* d_in, const int* in_sizes, int n_in,
                              void* d_out, int out_size, void* d_ws, size_t ws_size,
                              hipStream_t stream)
{
    (void)in_sizes; (void)n_in; (void)out_size; (void)ws_size;
    int ip = 0;
    auto nx = [&]() { return (const float*)d_in[ip++]; };
    auto readLayer = [&](LayerP& L) {
        L.b1 = nx(); L.b2 = nx(); L.bo = nx();
        L.ln1_b = nx(); L.ln1_g = nx(); L.ln2_b = nx(); L.ln2_g = nx();
        L.w1 = nx(); L.w2 = nx(); L.wo = nx(); L.wqkv = nx();
    };
    auto readTrans = [&](TransP& T) {
        T.be = nx(); T.cls = nx();
        readLayer(T.L[0]); readLayer(T.L[1]);
        T.we = nx();
    };
    auto readRes = [&](ResP& R) {
        R.b1 = nx(); R.b2 = nx(); R.be1 = nx(); R.be2 = nx();
        R.g1 = nx(); R.g2 = nx(); R.rb = nx(); R.rw = nx();
        R.w1 = nx(); R.w2 = nx();
    };

    const float* data = nx();
    const float* xs   = nx();
    // params (sorted dict keys): conv_b, conv_beta, conv_g, conv_w, embed1, l0, l1, lin1_b, lin1_w
    const float* conv_b = nx(); const float* conv_beta = nx();
    const float* conv_g = nx(); const float* conv_w = nx();
    ResP embed1; readRes(embed1);
    TransP l0[4];
    for (int i = 0; i < 4; ++i) readTrans(l0[i]);
    ResP l1_embed; readRes(l1_embed);
    const float* gsda_b = nx(); const float* gsda_beta = nx();
    const float* gsda_g = nx(); const float* gsda_w = nx();
    const float* lb = nx(); const float* lw = nx();
    TransP l1_trans; readTrans(l1_trans);
    const float* lin1_b = nx(); const float* lin1_w = nx();

    Ws w;
    float* f = (float*)d_ws;
    auto carve = [&](long n) { float* p = f; f += n; return p; };
    w.bufA = carve((long)BATCH * 2 * CH * NPT);
    w.bufB = carve((long)BATCH * 2 * CH * NPT);
    w.bufC = carve((long)BATCH * 2 * CH * NPT);
    w.t    = carve((long)BATCH * NTOK * CH);
    w.tn   = carve((long)BATCH * NTOK * CH);
    w.qkv  = carve((long)BATCH * NTOK * 3 * CH);
    w.attnO= carve((long)BATCH * NTOK * CH);
    w.h1   = carve((long)BATCH * NTOK * CH);
    w.mv   = carve(2 * 2 * CH);
    w.wr   = carve((long)BATCH * NPT);
    w.dv   = carve((long)BATCH * NPT);
    w.dwr  = carve((long)BATCH * NPT);
    w.dot0 = carve((long)BATCH * CH);
    w.dot1 = carve((long)BATCH * CH);
    w.w0l  = carve((long)BATCH * NPT);
    w.xwx  = carve((long)BATCH * 45);
    w.bufD = w.qkv;   // alias: (B,256,NPT) fits in qkv region; never live together

    float* logits_out = (float*)d_out;                  // (B, NPT)
    float* ehat_out   = logits_out + (long)BATCH * NPT; // (B, 9)
    float* resid_out  = ehat_out + (long)BATCH * 9;     // (B, NPT)

    const long totC = (long)BATCH * CH * NPT;
    const int ebC = (int)((totC + 255) / 256);

    // 1. out = relu(bn(conv1x1(data, conv_w, conv_b)))
    gemm(stream, conv_w, data, conv_b, 2, 0, w.bufA,
         CH, NPT, 4, 0, (long)4 * NPT, (long)CH * NPT, 4, NPT, NPT, 0, BATCH);
    k_bn_stats<<<CH, 256, 0, stream>>>(w.bufA, w.mv, BATCH, CH, NPT);
    k_bn_apply<<<ebC, 256, 0, stream>>>(w.bufA, w.mv, conv_g, conv_beta, w.bufA, CH, NPT, 1, totC);

    // 2. four transformer blocks (l0)
    run_transm(stream, w, w.bufA, w.bufB, l0[0]);
    run_transm(stream, w, w.bufB, w.bufA, l0[1]);
    run_transm(stream, w, w.bufA, w.bufB, l0[2]);
    run_transm(stream, w, w.bufB, w.bufA, l0[3]);

    // 3. trans_gd
    run_transm(stream, w, w.bufA, w.bufB, l1_trans);                          // x0 -> bufB
    run_resnet(stream, w, w.bufB, w.bufA, w.bufC, w.bufD, l1_embed, CH, CH);  // out0 -> bufA
    k_rowdot<<<(int)(((long)BATCH * NPT + 255) / 256), 256, 0, stream>>>(
        w.bufA, lw, lb, w.w0l, CH, NPT, (long)BATCH * NPT);
    k_cgns_w<<<BATCH, 256, 0, stream>>>(w.w0l, w.wr, w.dv, w.dwr, NPT);
    k_cgns_dots<<<BATCH * CH, 256, 0, stream>>>(w.bufA, w.wr, w.dwr, w.dot0, w.dot1, CH, NPT);
    k_cgns_combine<<<ebC, 256, 0, stream>>>(w.bufA, w.wr, w.dv, w.dwr, w.dot0, w.dot1,
                                            w.bufB, CH, NPT, totC);           // cat -> bufB
    gemm(stream, gsda_w, w.bufB, gsda_b, 2, 0, w.bufC,
         CH, NPT, 2 * CH, 0, (long)2 * CH * NPT, (long)CH * NPT, 2 * CH, NPT, NPT, 0, BATCH);
    k_bn_stats<<<CH, 256, 0, stream>>>(w.bufC, w.mv, BATCH, CH, NPT);
    k_bn_apply<<<ebC, 256, 0, stream>>>(w.bufC, w.mv, gsda_g, gsda_beta, w.bufC, CH, NPT, 1, totC);
    k_add<<<ebC, 256, 0, stream>>>(w.bufC, w.bufA, w.bufB, 0, totC);          // x1 -> bufB

    // 4. out1 = resnet_block(x1, embed1) : 128 -> 256 channels
    run_resnet(stream, w, w.bufB, w.bufA, w.bufC, w.bufD, embed1, CH, 2 * CH); // out1 -> bufA

    // 5. logits
    k_rowdot<<<(int)(((long)BATCH * NPT + 255) / 256), 256, 0, stream>>>(
        w.bufA, lin1_w, lin1_b, logits_out, 2 * CH, NPT, (long)BATCH * NPT);

    // 6. weighted 8-points -> e_hat
    k_xwx<<<BATCH, 256, 0, stream>>>(xs, logits_out, w.xwx, NPT);
    k_eigh9<<<1, 32, 0, stream>>>(w.xwx, ehat_out, BATCH);

    // 7. epipolar symmetric residual
    k_episym<<<(int)(((long)BATCH * NPT + 255) / 256), 256, 0, stream>>>(
        xs, ehat_out, resid_out, NPT, (long)BATCH * NPT);
}
